// ChannelAttention_LKA3D_86096914416208
// MI455X (gfx1250) — matrix operand order repaired and checked by reading, not compiled, not run on hardware
//
#include <hip/hip_runtime.h>

// ---------------- types ----------------
typedef __bf16  v16bf __attribute__((ext_vector_type(16)));
typedef __bf16  v8bf  __attribute__((ext_vector_type(8)));
typedef __bf16  v4bf  __attribute__((ext_vector_type(4)));
typedef float   v8f   __attribute__((ext_vector_type(8)));

#define Bq   4
#define Nq   16384
#define Cq   384
#define Hq   4
#define Dq   96
#define BNq  (Bq*Nq)
#define LDS_STRIDE 40   // elements per tile line (32 data + 8 pad), 80B, 16B aligned

// ---------------- CDNA5 async global->LDS (B128, 16B per lane) ----------------
__device__ __forceinline__ void async_b128(unsigned lds_off, const void* gptr) {
    asm volatile("global_load_async_to_lds_b128 %0, %1, off"
                 :: "v"(lds_off), "v"(gptr) : "memory");
}
__device__ __forceinline__ void wait_async0() {
    asm volatile("s_wait_asynccnt 0" ::: "memory");
}

// ---------------- WMMA fragment loaders (ISA 7.12.2 layouts) ----------------
// tile: [line][LDS_STRIDE] bf16 in LDS, K-contiguous per line (K=0..31 valid)
__device__ __forceinline__ v16bf load_a_frag(const __bf16* tile, int line0, int lane) {
    // A 16x32: lanes 0-15 -> M=lane, K {0..7,16..23}; lanes 16-31 -> M=lane-15? (lane&15), K {8..15,24..31}
    const __bf16* row = tile + (line0 + (lane & 15)) * LDS_STRIDE;
    int kb = (lane >> 4) << 3;                 // 0 or 8
    v8bf lo = *(const v8bf*)(row + kb);        // K kb..kb+7
    v8bf hi = *(const v8bf*)(row + kb + 16);   // K kb+16..kb+23
    v16bf f;
#pragma unroll
    for (int i = 0; i < 8; ++i) { f[i] = lo[i]; f[i + 8] = hi[i]; }
    return f;
}
__device__ __forceinline__ v16bf load_b_frag(const __bf16* tile, int line0, int lane) {
    // B 32x16: lanes 0-15 -> N=lane, K {0..7,16..23}; lanes 16-31 -> N=lane&15, K {8..15,24..31}
    const __bf16* row = tile + (line0 + (lane & 15)) * LDS_STRIDE + ((lane >> 4) << 3);
    v8bf lo = *(const v8bf*)(row);
    v8bf hi = *(const v8bf*)(row + 16);
    v16bf f;
#pragma unroll
    for (int i = 0; i < 8; ++i) { f[i] = lo[i]; f[i + 8] = hi[i]; }
    return f;
}

// 8 waves: wave computes 32(rows) x 64(cols) of a 128x128 tile; one K=32 step.
__device__ __forceinline__ void wave_mma_step(const __bf16* tA, const __bf16* tB,
                                              v8f acc[2][4], int wave, int lane) {
    const int wm = (wave & 3) * 32;
    const int wn = (wave >> 2) * 64;
    v16bf a[2], b[4];
#pragma unroll
    for (int i = 0; i < 2; ++i) a[i] = load_a_frag(tA, wm + i * 16, lane);
#pragma unroll
    for (int j = 0; j < 4; ++j) b[j] = load_b_frag(tB, wn + j * 16, lane);
#pragma unroll
    for (int i = 0; i < 2; ++i)
#pragma unroll
        for (int j = 0; j < 4; ++j)
            acc[i][j] = __builtin_amdgcn_wmma_f32_16x16x32_bf16(
                false, a[i], false, b[j], (short)0, acc[i][j], false, false);
}

// ---------------- K0: fp32 -> bf16 convert ----------------
__global__ void convert_kernel(const float* __restrict__ x, __bf16* __restrict__ xb) {
    size_t i = ((size_t)blockIdx.x * blockDim.x + threadIdx.x) * 4;
    float4 v = *(const float4*)(x + i);
    v4bf o; o[0] = (__bf16)v.x; o[1] = (__bf16)v.y; o[2] = (__bf16)v.z; o[3] = (__bf16)v.w;
    *(v4bf*)(xb + i) = o;
}

// ---------------- K1: S_b = X_b^T X_b  (split-K over tokens, atomic f32) ----------------
#define NSPLIT 16
__global__ void ks_kernel(const __bf16* __restrict__ xb, float* __restrict__ S) {
    __shared__ __align__(16) __bf16 tileA[128 * LDS_STRIDE];
    __shared__ __align__(16) __bf16 tileB[128 * LDS_STRIDE];
    const int b     = blockIdx.y;
    const int tile  = blockIdx.x % 9;           // 3x3 c-tiles of 128
    const int chunk = blockIdx.x / 9;
    const int cA0 = (tile / 3) * 128;
    const int cB0 = (tile % 3) * 128;
    const __bf16* X = xb + (size_t)b * Nq * Cq;
    const int n0   = chunk * (Nq / NSPLIT);
    const int nEnd = n0 + (Nq / NSPLIT);
    const int tid = threadIdx.x, lane = tid & 31, wave = tid >> 5;
    v8f acc[2][4] = {};
    for (int n = n0; n < nEnd; n += 32) {
        __syncthreads();
        // load 32 rows x 128 cols (x2) coalesced, store transposed: tile[c][k=n-row]
        const int r = tid >> 3, seg = tid & 7;           // row 0..31, col-seg 0..7
        const __bf16* sA = X + (size_t)(n + r) * Cq + cA0 + seg * 16;
        const __bf16* sB = X + (size_t)(n + r) * Cq + cB0 + seg * 16;
        v8bf a0 = *(const v8bf*)(sA);     v8bf a1 = *(const v8bf*)(sA + 8);
        v8bf b0 = *(const v8bf*)(sB);     v8bf b1 = *(const v8bf*)(sB + 8);
#pragma unroll
        for (int j = 0; j < 8; ++j) {
            tileA[(seg * 16 + j)     * LDS_STRIDE + r] = a0[j];
            tileA[(seg * 16 + 8 + j) * LDS_STRIDE + r] = a1[j];
            tileB[(seg * 16 + j)     * LDS_STRIDE + r] = b0[j];
            tileB[(seg * 16 + 8 + j) * LDS_STRIDE + r] = b1[j];
        }
        __syncthreads();
        wave_mma_step(tileA, tileB, acc, wave, lane);
    }
    float* Sb = S + (size_t)b * Cq * Cq;
    const int wm = (wave & 3) * 32, wn = (wave >> 2) * 64;
    const int col = lane & 15, rh = lane >> 4;
#pragma unroll
    for (int i = 0; i < 2; ++i)
#pragma unroll
        for (int j = 0; j < 4; ++j)
#pragma unroll
            for (int r = 0; r < 8; ++r) {
                int row = cA0 + wm + i * 16 + rh * 8 + r;
                int c   = cB0 + wn + j * 16 + col;
                atomicAdd(&Sb[(size_t)row * Cq + c], acc[i][j][r]);
            }
}

// ---------------- mid kernels (tiny fp32 work) ----------------
// T = Wsub(rows rowOff..rowOff+383) @ S_b ; layout (b, r, j)
__global__ void midT_kernel(const float* __restrict__ Wqkv, const float* __restrict__ S,
                            float* __restrict__ T, int rowOff) {
    int idx = blockIdx.x * 256 + threadIdx.x;            // 4*384*384
    int b = idx / (Cq * Cq), rem = idx % (Cq * Cq);
    int r = rem / Cq, j = rem % Cq;
    const float* w = Wqkv + (size_t)(rowOff + r) * Cq;
    const float* s = S + (size_t)b * Cq * Cq;
    float acc = 0.f;
    for (int i = 0; i < Cq; ++i) acc += w[i] * s[(size_t)i * Cq + j];
    T[idx] = acc;
}
// norms: nrm[b][0..383]=||q_c||, [384..767]=||k_d||
__global__ void midNorm_kernel(const float* __restrict__ Tq, const float* __restrict__ Tk,
                               const float* __restrict__ Wqkv, float* __restrict__ nrm) {
    int idx = blockIdx.x * 256 + threadIdx.x;            // 4*768
    if (idx >= Bq * 2 * Cq) return;
    int b = idx / (2 * Cq), r = idx % (2 * Cq);
    const float* T = (r < Cq ? Tq : Tk) + (size_t)b * Cq * Cq + (size_t)(r % Cq) * Cq;
    const float* w = Wqkv + (size_t)r * Cq;              // q rows 0..383, k rows 384..767
    float acc = 0.f;
    for (int i = 0; i < Cq; ++i) acc += T[i] * w[i];
    nrm[idx] = sqrtf(fmaxf(acc, 0.f));
}
// G[b,h,c,d] = Tq[b,h*96+c,:] . Wk[h*96+d,:]
__global__ void midG_kernel(const float* __restrict__ Tq, const float* __restrict__ Wqkv,
                            float* __restrict__ G) {
    int idx = blockIdx.x * 256 + threadIdx.x;            // 4*4*96*96
    int d = idx % Dq, c = (idx / Dq) % Dq, h = (idx / (Dq * Dq)) % Hq, b = idx / (Dq * Dq * Hq);
    const float* t = Tq + ((size_t)b * Cq + h * Dq + c) * Cq;
    const float* w = Wqkv + (size_t)(Cq + h * Dq + d) * Cq;
    float acc = 0.f;
    for (int i = 0; i < Cq; ++i) acc += t[i] * w[i];
    G[idx] = acc;
}
// softmax row (b,h,c) over d
__global__ void midAttn_kernel(const float* __restrict__ G, const float* __restrict__ nrm,
                               const float* __restrict__ temp, float* __restrict__ attn) {
    int idx = blockIdx.x * 256 + threadIdx.x;            // 4*4*96 rows
    if (idx >= Bq * Hq * Dq) return;
    int c = idx % Dq, h = (idx / Dq) % Hq, b = idx / (Dq * Hq);
    const float* g  = G + (size_t)idx * Dq;
    const float* nq = nrm + b * 2 * Cq + h * Dq;
    const float* nk = nrm + b * 2 * Cq + Cq + h * Dq;
    const float eps = 1e-12f;
    const float qn = fmaxf(nq[c], eps);
    const float tp = temp[h];
    float mx = -3.4e38f;
    for (int d = 0; d < Dq; ++d)
        mx = fmaxf(mx, g[d] * tp / (qn * fmaxf(nk[d], eps)));
    float sum = 0.f;
    for (int d = 0; d < Dq; ++d)
        sum += __expf(g[d] * tp / (qn * fmaxf(nk[d], eps)) - mx);
    float inv = 1.f / sum;
    for (int d = 0; d < Dq; ++d)
        attn[(size_t)idx * Dq + d] = __expf(g[d] * tp / (qn * fmaxf(nk[d], eps)) - mx) * inv;
}
// Mcat[b][o][h*96+d] = sum_c Wout[o][h*96+c]*attn[b,h,c,d]
__global__ void midM_kernel(const float* __restrict__ Wout, const float* __restrict__ attn,
                            float* __restrict__ Mcat) {
    int idx = blockIdx.x * 256 + threadIdx.x;            // 4*384*384
    int j = idx % Cq, o = (idx / Cq) % Cq, b = idx / (Cq * Cq);
    int h = j / Dq, d = j % Dq;
    const float* w = Wout + (size_t)o * Cq + h * Dq;
    const float* a = attn + (((size_t)b * Hq + h) * Dq) * Dq + d;
    float acc = 0.f;
    for (int c = 0; c < Dq; ++c) acc += w[c] * a[(size_t)c * Dq];
    Mcat[idx] = acc;
}
// Pt[b][o][k] = sum_j Mcat[b][o][j] * Wv[j][k]   (bf16 out)
__global__ void midP_kernel(const float* __restrict__ Mcat, const float* __restrict__ Wqkv,
                            __bf16* __restrict__ Pt) {
    int idx = blockIdx.x * 256 + threadIdx.x;            // 4*384*384
    int k = idx % Cq, o = (idx / Cq) % Cq, b = idx / (Cq * Cq);
    const float* m = Mcat + ((size_t)b * Cq + o) * Cq;
    float acc = 0.f;
    for (int j = 0; j < Cq; ++j) acc += m[j] * Wqkv[(size_t)(2 * Cq + j) * Cq + k];
    Pt[idx] = (__bf16)acc;
}

// ---------------- K2: out_b = X_b @ Pt_b^T + bias  (async LDS fills) ----------------
__global__ void kout_kernel(const __bf16* __restrict__ xb, const __bf16* __restrict__ Pt,
                            const float* __restrict__ bias, float* __restrict__ out) {
    __shared__ __align__(16) __bf16 tileA[128 * LDS_STRIDE];
    __shared__ __align__(16) __bf16 tileB[128 * LDS_STRIDE];
    const int b   = blockIdx.z;
    const int m0  = blockIdx.x * 128;
    const int n0c = blockIdx.y * 128;
    const __bf16* X = xb + (size_t)b * Nq * Cq;
    const __bf16* P = Pt + (size_t)b * Cq * Cq;
    float* O = out + (size_t)b * Nq * Cq;
    const int tid = threadIdx.x, lane = tid & 31, wave = tid >> 5;
    const unsigned ldsA = (unsigned)(unsigned long long)(void*)tileA;
    const unsigned ldsB = (unsigned)(unsigned long long)(void*)tileB;
    v8f acc[2][4] = {};
    for (int k0 = 0; k0 < Cq; k0 += 32) {
        __syncthreads();   // previous compute done before overwriting LDS
#pragma unroll
        for (int i = 0; i < 2; ++i) {
            int idx  = tid + i * 256;         // 512 chunks x 16B = two 8KB tiles
            int line = idx >> 2;
            int part = idx & 3;
            unsigned loff = (unsigned)(line * LDS_STRIDE + part * 8) * 2u; // bytes
            async_b128(ldsA + loff, X + (size_t)(m0 + line) * Cq + k0 + part * 8);
            async_b128(ldsB + loff, P + (size_t)(n0c + line) * Cq + k0 + part * 8);
        }
        wait_async0();
        __syncthreads();
        wave_mma_step(tileA, tileB, acc, wave, lane);
    }
    const int wm = (wave & 3) * 32, wn = (wave >> 2) * 64;
    const int col = lane & 15, rh = lane >> 4;
#pragma unroll
    for (int i = 0; i < 2; ++i)
#pragma unroll
        for (int j = 0; j < 4; ++j) {
            int o = n0c + wn + j * 16 + col;
            float bv = bias[o];
#pragma unroll
            for (int r = 0; r < 8; ++r) {
                int row = m0 + wm + i * 16 + rh * 8 + r;
                O[(size_t)row * Cq + o] = acc[i][j][r] + bv;
            }
        }
}

// ---------------- host ----------------
extern "C" void kernel_launch(void* const* d_in, const int* in_sizes, int n_in,
                              void* d_out, int out_size, void* d_ws, size_t ws_size,
                              hipStream_t stream) {
    (void)in_sizes; (void)n_in; (void)out_size; (void)ws_size;
    const float* x    = (const float*)d_in[0];
    const float* Wqkv = (const float*)d_in[1];
    const float* temp = (const float*)d_in[2];
    const float* Wout = (const float*)d_in[3];
    const float* bout = (const float*)d_in[4];
    float* out = (float*)d_out;

    char* ws = (char*)d_ws;
    __bf16* xb  = (__bf16*)(ws);                 // 50,331,648 B
    float*  S   = (float*)(ws + 50331648);       //  2,359,296 B
    float*  Tq  = (float*)(ws + 52690944);       //  2,359,296 B
    float*  Tk  = (float*)(ws + 55050240);       //  2,359,296 B
    float*  nrm = (float*)(ws + 57409536);       //     12,288 B
    float*  G   = (float*)(ws + 57421824);       //    589,824 B
    float*  att = (float*)(ws + 58011648);       //    589,824 B
    float*  Mc  = (float*)(ws + 58601472);       //  2,359,296 B
    __bf16* Pt  = (__bf16*)(ws + 60960768);      //  1,179,648 B  (total ~62.1 MB)

    hipMemsetAsync(S, 0, (size_t)Bq * Cq * Cq * sizeof(float), stream);
    convert_kernel<<<(BNq * Cq) / 1024, 256, 0, stream>>>(x, xb);
    ks_kernel<<<dim3(9 * NSPLIT, Bq), 256, 0, stream>>>(xb, S);
    midT_kernel<<<(Bq * Cq * Cq) / 256, 256, 0, stream>>>(Wqkv, S, Tq, 0);
    midT_kernel<<<(Bq * Cq * Cq) / 256, 256, 0, stream>>>(Wqkv, S, Tk, Cq);
    midNorm_kernel<<<(Bq * 2 * Cq + 255) / 256, 256, 0, stream>>>(Tq, Tk, Wqkv, nrm);
    midG_kernel<<<(Bq * Hq * Dq * Dq) / 256, 256, 0, stream>>>(Tq, Wqkv, G);
    midAttn_kernel<<<(Bq * Hq * Dq + 255) / 256, 256, 0, stream>>>(G, nrm, temp, att);
    midM_kernel<<<(Bq * Cq * Cq) / 256, 256, 0, stream>>>(Wout, att, Mc);
    midP_kernel<<<(Bq * Cq * Cq) / 256, 256, 0, stream>>>(Mc, Wqkv, Pt);
    kout_kernel<<<dim3(Nq / 128, Cq / 128, Bq), 256, 0, stream>>>(xb, Pt, bout, out);
}
